// LSTMClassifier_7189775254158
// MI455X (gfx1250) — matrix operand order repaired and checked by reading, not compiled, not run on hardware
//
#include <hip/hip_runtime.h>
#include <hip/hip_bf16.h>
#include <math.h>

typedef __attribute__((ext_vector_type(16))) _Float16 v16h;
typedef __attribute__((ext_vector_type(8)))  float    v8f;

#define LSTM_B   128
#define LSTM_T   256
#define LSTM_IN  512
#define LSTM_H   512
#define LSTM_4H  2048
#define LSTM_C   64

__device__ __forceinline__ float sigmoidf_(float x) {
    return 1.0f / (1.0f + __expf(-x));
}

// ---------------------------------------------------------------------------
// WMMA fragment loaders (layouts per CDNA5 ISA 7.12.2, 16-bit A 16x32 / B 32x16)
// ---------------------------------------------------------------------------

// A fragment: rowp = &A[row][kbase] (row-major; per-lane row resolved by caller)
__device__ __forceinline__ v16h load_a_frag(const _Float16* __restrict__ rowp,
                                            unsigned lane) {
    const unsigned g = (lane >> 4) * 8u;   // lane group 0/1 -> K sub-block
    v16h a;
#pragma unroll
    for (int h = 0; h < 8; ++h) a[h] = rowp[g + h];
#pragma unroll
    for (int h = 0; h < 8; ++h) a[8 + h] = rowp[16 + g + h];
    return a;
}

// B fragment from PRE-TRANSPOSED weights Wt[K][N] (f16, row-major, ld = N):
// lane%16 = column n0+lane%16, elements = K rows in table order (8h+s, 16+8h+s).
__device__ __forceinline__ v16h load_bt_frag(const _Float16* __restrict__ Wt,
                                             int ldn, int n0, int kbase,
                                             unsigned lane) {
    const _Float16* p = Wt + (size_t)kbase * ldn + n0 + (lane & 15u);
    const unsigned g = (lane >> 4) * 8u;
    v16h b;
#pragma unroll
    for (int h = 0; h < 8; ++h) b[h] = p[(size_t)(g + h) * ldn];
#pragma unroll
    for (int h = 0; h < 8; ++h) b[8 + h] = p[(size_t)(16 + g + h) * ldn];
    return b;
}

// ---------------------------------------------------------------------------
// Small helper kernels
// ---------------------------------------------------------------------------

// x [B,T,IN] f32 -> xt [T,B,IN] f16  (t-major activations)
__global__ void cvt_x_tmajor(const float* __restrict__ x,
                             _Float16* __restrict__ xt, int n) {
    int idx = blockIdx.x * blockDim.x + threadIdx.x;
    if (idx < n) {
        const int IN = LSTM_IN, T = LSTM_T, B = LSTM_B;
        int i   = idx % IN;
        int row = idx / IN;          // b*T + t
        int t   = row % T;
        int b   = row / T;
        xt[((size_t)t * B + b) * IN + i] = (_Float16)x[idx];
    }
}

// W [N,K] f32  ->  Wt [K,N] f16   (write-coalesced)
__global__ void cvt_transpose(const float* __restrict__ W,
                              _Float16* __restrict__ Wt,
                              int N, int K) {
    int idx = blockIdx.x * blockDim.x + threadIdx.x;
    if (idx < N * K) {
        int k = idx / N;
        int n = idx - k * N;
        Wt[idx] = (_Float16)W[(size_t)n * K + k];
    }
}

__global__ void bias_sum(const float* __restrict__ a,
                         const float* __restrict__ b,
                         float* __restrict__ out, int n) {
    int i = blockIdx.x * blockDim.x + threadIdx.x;
    if (i < n) out[i] = a[i] + b[i];
}

__global__ void zero_h16_c32(_Float16* __restrict__ h,
                             float* __restrict__ c, int n) {
    int i = blockIdx.x * blockDim.x + threadIdx.x;
    if (i < n) { h[i] = (_Float16)0.0f; c[i] = 0.0f; }
}

// ---------------------------------------------------------------------------
// Big GEMM: O[M,N](f16) = A[M,K](f16) @ Wt[K,N](f16)   (input projections)
// One wave -> 16x64 strip (4 N-tiles sharing one A fragment per K-step).
// 4 warps/block share the same A rows (WGP$ reuse).  Grid = M/16*N/64/4.
// ---------------------------------------------------------------------------
__global__ void __launch_bounds__(128)
gemm_f16_tn(const _Float16* __restrict__ A,
            const _Float16* __restrict__ Wt,
            _Float16* __restrict__ O,
            int M, int N, int K) {
    const unsigned lane = threadIdx.x & 31u;
    const unsigned warp = threadIdx.x >> 5;
    const int nstrips = N >> 6;
    const int tile = blockIdx.x * 4 + (int)warp;
    const int mt = tile / nstrips;
    const int nt = tile % nstrips;
    const int m0 = mt * 16, n0 = nt * 64;
    const int arow = m0 + (int)(lane & 15u);

    v8f acc[4] = {};
    const _Float16* arp = A + (size_t)arow * K;
    for (int k = 0; k < K; k += 32) {
        v16h af = load_a_frag(arp + k, lane);
#pragma unroll
        for (int j = 0; j < 4; ++j) {
            v16h bf = load_bt_frag(Wt, N, n0 + j * 16, k, lane);
            acc[j] = __builtin_amdgcn_wmma_f32_16x16x32_f16(
                false, af, false, bf, (short)0, acc[j], false, false);
        }
    }

    const int orow0 = m0 + (int)((lane >> 4) << 3);
    const int ocol0 = n0 + (int)(lane & 15u);
#pragma unroll
    for (int j = 0; j < 4; ++j)
#pragma unroll
        for (int r = 0; r < 8; ++r)
            O[(size_t)(orow0 + r) * N + ocol0 + j * 16] = (_Float16)acc[j][r];
}

// ---------------------------------------------------------------------------
// Fused LSTM step (t-major): gates = xpt + bias + hin @ WhhT, then the cell
// update.  Each wave owns a 16(batch) x 16(hidden) patch and computes all
// four gate tiles (4 WMMA chains sharing the A fragment).  The h-sequence
// buffer doubles as the h state: hin = slab t-1, hout = slab t.
// xproj gate values are preloaded BEFORE the K loop so their latency hides
// under the WMMA chain.  Grid: (8 mtiles * 32 utiles)/4 = 64 blocks of 128.
// ---------------------------------------------------------------------------
__global__ void __launch_bounds__(128)
lstm_step(const _Float16* __restrict__ xpt,    // [B, 4H] slab for this t
          const _Float16* __restrict__ WhhT,   // [H, 4H]
          const float*    __restrict__ bias,   // [4H]  (b_ih+b_hh)
          const _Float16* __restrict__ hin,    // [B, H] slab t-1 (or zeros)
          _Float16*       __restrict__ hout,   // [B, H] slab t of hseq
          float*          __restrict__ cbuf)   // [B, H]
{
    const int H = LSTM_H, G = LSTM_4H;
    const unsigned lane = threadIdx.x & 31u;
    const unsigned warp = threadIdx.x >> 5;
    const int tile = blockIdx.x * 4 + (int)warp;   // 0..255
    const int mt = tile >> 5;                      // batch tile 0..7
    const int ut = tile & 31;                      // hidden tile 0..31
    const int m0 = mt * 16, u0 = ut * 16;
    const int arow = m0 + (int)(lane & 15u);

    const int b0row = m0 + (int)((lane >> 4) << 3);
    const int u = u0 + (int)(lane & 15u);

    // Preload xproj gate inputs (issued early; consumed after the WMMA chain)
    float xg[4][8];
#pragma unroll
    for (int j = 0; j < 4; ++j)
#pragma unroll
        for (int r = 0; r < 8; ++r)
            xg[j][r] = (float)xpt[(size_t)(b0row + r) * G + j * H + u];

    const float bi_ = bias[0 * H + u];
    const float bf_ = bias[1 * H + u];
    const float bg_ = bias[2 * H + u];
    const float bo_ = bias[3 * H + u];

    v8f acc[4] = {};
    const _Float16* arp = hin + (size_t)arow * H;
    for (int k = 0; k < H; k += 32) {
        v16h af = load_a_frag(arp + k, lane);
#pragma unroll
        for (int j = 0; j < 4; ++j) {
            v16h bf = load_bt_frag(WhhT, G, j * H + u0, k, lane);
            acc[j] = __builtin_amdgcn_wmma_f32_16x16x32_f16(
                false, af, false, bf, (short)0, acc[j], false, false);
        }
    }

#pragma unroll
    for (int r = 0; r < 8; ++r) {
        const int b = b0row + r;
        float gi = acc[0][r] + bi_ + xg[0][r];
        float gf = acc[1][r] + bf_ + xg[1][r];
        float gg = acc[2][r] + bg_ + xg[2][r];
        float go = acc[3][r] + bo_ + xg[3][r];
        float iv = sigmoidf_(gi);
        float fv = sigmoidf_(gf);
        float gv = tanhf(gg);
        float ov = sigmoidf_(go);
        const size_t ci = (size_t)b * H + u;
        float cn = fv * cbuf[ci] + iv * gv;
        cbuf[ci] = cn;
        hout[ci] = (_Float16)(ov * tanhf(cn));
    }
}

// ---------------------------------------------------------------------------
// Head: out[b,t,c] = sigmoid( h1[t,b,:] @ Wlin[c,:] + b_lin[c] ), t < T-1.
// h1 is t-major [T,B,H]; WlT pre-transposed [H,C].
// One wave -> 16 output rows x all 64 cols.  2040 waves -> 510 blocks.
// ---------------------------------------------------------------------------
__global__ void __launch_bounds__(128)
final_linear(const _Float16* __restrict__ h1,   // [T, B, H]
             const _Float16* __restrict__ WlT,  // [H, C]
             const float*    __restrict__ bl,   // [C]
             float*          __restrict__ out)  // [B, T-1, C]
{
    const int H = LSTM_H, B = LSTM_B, Tm1 = LSTM_T - 1, C = LSTM_C;
    const unsigned lane = threadIdx.x & 31u;
    const unsigned warp = threadIdx.x >> 5;
    const int mt = blockIdx.x * 4 + (int)warp;   // 0..2039
    const int m0 = mt * 16;

    const int m = m0 + (int)(lane & 15u);        // output row: b*(T-1)+t
    const int b = m / Tm1;
    const int t = m - b * Tm1;
    const _Float16* arp = h1 + (size_t)(t * B + b) * H;

    v8f acc[4] = {};
    for (int k = 0; k < H; k += 32) {
        v16h af = load_a_frag(arp + k, lane);
#pragma unroll
        for (int j = 0; j < 4; ++j) {
            v16h bf = load_bt_frag(WlT, C, j * 16, k, lane);
            acc[j] = __builtin_amdgcn_wmma_f32_16x16x32_f16(
                false, af, false, bf, (short)0, acc[j], false, false);
        }
    }

    const int orow0 = m0 + (int)((lane >> 4) << 3);
    const int ocol0 = (int)(lane & 15u);
#pragma unroll
    for (int j = 0; j < 4; ++j) {
        const float bc = bl[ocol0 + j * 16];
#pragma unroll
        for (int r = 0; r < 8; ++r)
            out[(size_t)(orow0 + r) * C + ocol0 + j * 16] =
                sigmoidf_(acc[j][r] + bc);
    }
}

// ---------------------------------------------------------------------------
// Host orchestration
// ---------------------------------------------------------------------------
extern "C" void kernel_launch(void* const* d_in, const int* in_sizes, int n_in,
                              void* d_out, int out_size, void* d_ws, size_t ws_size,
                              hipStream_t stream) {
    const int B = LSTM_B, T = LSTM_T, IN = LSTM_IN, H = LSTM_H, C = LSTM_C;
    const int M  = B * T;       // 32768
    const int G  = 4 * H;       // 2048

    const float* x    = (const float*)d_in[0];
    const float* Wih0 = (const float*)d_in[3];
    const float* Whh0 = (const float*)d_in[4];
    const float* bih0 = (const float*)d_in[5];
    const float* bhh0 = (const float*)d_in[6];
    const float* Wih1 = (const float*)d_in[7];
    const float* Whh1 = (const float*)d_in[8];
    const float* bih1 = (const float*)d_in[9];
    const float* bhh1 = (const float*)d_in[10];
    const float* Wlin = (const float*)d_in[11];
    const float* blin = (const float*)d_in[12];

    char* ws = (char*)d_ws;
    size_t off = 0;
    auto take = [&](size_t bytes) -> char* {
        char* p = ws + off;
        off = (off + bytes + 255) & ~(size_t)255;
        return p;
    };

    _Float16* x16    = (_Float16*)take((size_t)M * IN * 2);  // [T,B,IN]
    _Float16* xproj  = (_Float16*)take((size_t)M * G * 2);   // [T,B,4H]
    _Float16* h0seq  = (_Float16*)take((size_t)M * H * 2);   // [T,B,H]
    _Float16* h1seq  = (_Float16*)take((size_t)M * H * 2);   // [T,B,H]
    _Float16* wih0T  = (_Float16*)take((size_t)IN * G * 2);  // [IN, 4H]
    _Float16* whh0T  = (_Float16*)take((size_t)H * G * 2);   // [H, 4H]
    _Float16* wih1T  = (_Float16*)take((size_t)H * G * 2);   // [H, 4H]
    _Float16* whh1T  = (_Float16*)take((size_t)H * G * 2);   // [H, 4H]
    _Float16* wlinT  = (_Float16*)take((size_t)H * C * 2);   // [H, C]
    float*    bias0  = (float*)take((size_t)G * 4);
    float*    bias1  = (float*)take((size_t)G * 4);
    _Float16* hzero  = (_Float16*)take((size_t)B * H * 2);   // zero h slab
    float*    cbuf   = (float*)take((size_t)B * H * 4);
    (void)ws_size; (void)in_sizes; (void)n_in; (void)out_size;

    // --- precision conversion + t-major transpose + weight transposes ---
    {
        int n = M * IN;
        cvt_x_tmajor<<<(n + 255) / 256, 256, 0, stream>>>(x, x16, n);
        n = G * IN;
        cvt_transpose<<<(n + 255) / 256, 256, 0, stream>>>(Wih0, wih0T, G, IN);
        n = G * H;
        cvt_transpose<<<(n + 255) / 256, 256, 0, stream>>>(Whh0, whh0T, G, H);
        cvt_transpose<<<(n + 255) / 256, 256, 0, stream>>>(Wih1, wih1T, G, H);
        cvt_transpose<<<(n + 255) / 256, 256, 0, stream>>>(Whh1, whh1T, G, H);
        n = C * H;
        cvt_transpose<<<(n + 255) / 256, 256, 0, stream>>>(Wlin, wlinT, C, H);
        bias_sum<<<(G + 255) / 256, 256, 0, stream>>>(bih0, bhh0, bias0, G);
        bias_sum<<<(G + 255) / 256, 256, 0, stream>>>(bih1, bhh1, bias1, G);
    }

    const int gemm_blocks = (M / 16) * (G / 64) / 4;   // 16384
    const int step_blocks = ((B / 16) * (H / 16)) / 4; // 64
    const int nBH = B * H;

    // --- layer 0 ---
    gemm_f16_tn<<<gemm_blocks, 128, 0, stream>>>(x16, wih0T, xproj, M, G, IN);
    zero_h16_c32<<<(nBH + 255) / 256, 256, 0, stream>>>(hzero, cbuf, nBH);
    for (int t = 0; t < T; ++t) {
        const _Float16* hin = (t == 0) ? hzero : h0seq + (size_t)(t - 1) * B * H;
        lstm_step<<<step_blocks, 128, 0, stream>>>(
            xproj + (size_t)t * B * G, whh0T, bias0,
            hin, h0seq + (size_t)t * B * H, cbuf);
    }

    // --- layer 1 (reuse xproj buffer) ---
    gemm_f16_tn<<<gemm_blocks, 128, 0, stream>>>(h0seq, wih1T, xproj, M, G, H);
    zero_h16_c32<<<(nBH + 255) / 256, 256, 0, stream>>>(hzero, cbuf, nBH);
    for (int t = 0; t < T; ++t) {
        const _Float16* hin = (t == 0) ? hzero : h1seq + (size_t)(t - 1) * B * H;
        lstm_step<<<step_blocks, 128, 0, stream>>>(
            xproj + (size_t)t * B * G, whh1T, bias1,
            hin, h1seq + (size_t)t * B * H, cbuf);
    }

    // --- head ---
    const int head_blocks = (B * (T - 1) / 16) / 4;  // 510
    final_linear<<<head_blocks, 128, 0, stream>>>(h1seq, wlinT, blin,
                                                  (float*)d_out);
}
